// HardcodedAttention_22634477650241
// MI455X (gfx1250) — matrix operand order repaired and checked by reading, not compiled, not run on hardware
//
#include <hip/hip_runtime.h>
#include <hip/hip_bf16.h>

typedef _Float16 half_t;
typedef __attribute__((ext_vector_type(16))) _Float16 v16h;
typedef __attribute__((ext_vector_type(8)))  _Float16 v8h;
typedef __attribute__((ext_vector_type(8)))  float    v8f;
typedef __attribute__((ext_vector_type(4)))  float    v4f;

#define B_    4
#define S_    2048
#define HID_  1024
#define NOUT_ 1024
#define EPS_  1e-6f

// Byte strides for the packed-B fragment walk (compile-time constants so the
// hot loop is pointer += constant with immediate load offsets).
#define B_KT_STRIDE   65536   // 64 n-tiles * 32 lanes * 32 B per K-tile step
#define B_NT_STRIDE   1024    // one n-tile = 32 lanes * 32 B

// Workspace layout (bytes). All offsets are multiples of 32B for vector loads.
static constexpr size_t OFF_WV16 = 0;                                   // Wv packed f16:   2 MB
static constexpr size_t OFF_VRM  = OFF_WV16 + (size_t)HID_*NOUT_*2;     // value row-major: 16 MB
static constexpr size_t OFF_VPK  = OFF_VRM  + (size_t)B_*S_*NOUT_*2;    // value packed:    16 MB
static constexpr size_t OFF_A16  = OFF_VPK  + (size_t)B_*S_*NOUT_*2;    // masked attn f16: 32 MB
static constexpr size_t OFF_INV  = OFF_A16  + (size_t)B_*S_*S_*2;       // inv row sums:    32 KB

// ---------------------------------------------------------------------------
// Repack Wv (f32 row-major [K=1024][N=1024]) into WMMA B-fragment layout:
// dst[(((kt*64 + nt)*32 + lane)*16 + e)] = Wv[kt*32 + (lane<16?0:8) + (e&7) + (e&8?16:0)][nt*16 + lane%16]
// ---------------------------------------------------------------------------
__global__ void __launch_bounds__(256) repack_wv_kernel(const float* __restrict__ Wv,
                                                        half_t* __restrict__ dst) {
    int id   = blockIdx.x * 256 + threadIdx.x;   // HID_*NOUT_ total
    int e    = id & 15;
    int lane = (id >> 4) & 31;
    int tile = id >> 9;
    int nt   = tile & 63;        // NOUT_/16
    int kt   = tile >> 6;        // HID_/32
    int k    = kt * 32 + ((lane < 16) ? 0 : 8) + (e & 7) + ((e & 8) << 1);
    int n    = nt * 16 + (lane & 15);
    dst[id] = (half_t)Wv[(size_t)k * NOUT_ + n];
}

// ---------------------------------------------------------------------------
// Repack value (f16 row-major [B*S][NOUT]) into per-batch B-fragment layout.
// ---------------------------------------------------------------------------
__global__ void __launch_bounds__(256) repack_val_kernel(const half_t* __restrict__ src,
                                                         half_t* __restrict__ dst) {
    int id   = blockIdx.x * 256 + threadIdx.x;   // B_*S_*NOUT_ total
    int e    = id & 15;
    int lane = (id >> 4) & 31;
    int t    = id >> 9;
    int nt   = t & 63;           // NOUT_/16
    int kt   = (t >> 6) & 63;    // S_/32
    int b    = t >> 12;
    int k    = kt * 32 + ((lane < 16) ? 0 : 8) + (e & 7) + ((e & 8) << 1);
    int n    = nt * 16 + (lane & 15);
    dst[id] = src[((size_t)b * S_ + k) * NOUT_ + n];
}

// ---------------------------------------------------------------------------
// Causal mask + f16 convert + inverse row sums. One block per (b,q) row.
// ---------------------------------------------------------------------------
__global__ void __launch_bounds__(256) mask_rowsum_kernel(const float* __restrict__ attn,
                                                          half_t* __restrict__ a16,
                                                          float* __restrict__ inv) {
    int row = blockIdx.x;          // b*S + q
    int q   = row & (S_ - 1);
    const float* src = attn + (size_t)row * S_;
    half_t*      dst = a16  + (size_t)row * S_;
    float sum = 0.f;
    for (int k = threadIdx.x; k < S_; k += 256) {
        float v = (k <= q) ? src[k] : 0.f;
        dst[k] = (half_t)v;
        sum += v;
    }
    __shared__ float red[256];
    red[threadIdx.x] = sum;
    __syncthreads();
    for (int off = 128; off > 0; off >>= 1) {
        if (threadIdx.x < off) red[threadIdx.x] += red[threadIdx.x + off];
        __syncthreads();
    }
    if (threadIdx.x == 0) inv[row] = 1.0f / (EPS_ + red[0]);
}

// ---------------------------------------------------------------------------
// GEMM1: value = hidden(f32, converted in-register) @ Wv(f16 packed) + bv.
// Wave computes a 16x64 strip (1 M-tile x 4 N-tiles). 2-stage software
// pipeline: next iteration's loads are issued before this iteration's WMMAs.
// ---------------------------------------------------------------------------
__global__ void __launch_bounds__(256) gemm_value_kernel(const float* __restrict__ hid,
                                                         const half_t* __restrict__ wpk,
                                                         const float* __restrict__ bv,
                                                         half_t* __restrict__ vrm) {
    const int lane = threadIdx.x & 31;
    const int wave = threadIdx.x >> 5;
    const int mt   = blockIdx.x;               // 0..(B*S/16 - 1)
    const int ntg  = blockIdx.y * 8 + wave;    // 0..15
    const int ntb  = ntg * 4;                  // base N-tile
    const int hi16 = (lane < 16) ? 0 : 1;
    const int row  = mt * 16 + (lane & 15);

    // A: f32 row, per kt the lane reads bytes [0,16) [16,32) [64,80) [80,96)
    // from (row base + hi16*32 + kt*128).
    const char* pa = (const char*)(hid + (size_t)row * HID_) + hi16 * 32;
    // B: packed fragments; per kt stride B_KT_STRIDE, fragment i at +i*B_NT_STRIDE.
    const char* pb = (const char*)wpk + (size_t)(ntb * 32 + lane) * 32;

    v8f acc[4] = {};

    // Prologue: stage kt = 0.
    v4f f0 = *(const v4f*)(pa +  0);
    v4f f1 = *(const v4f*)(pa + 16);
    v4f f2 = *(const v4f*)(pa + 64);
    v4f f3 = *(const v4f*)(pa + 80);
    v16h b0 = *(const v16h*)(pb + 0 * B_NT_STRIDE);
    v16h b1 = *(const v16h*)(pb + 1 * B_NT_STRIDE);
    v16h b2 = *(const v16h*)(pb + 2 * B_NT_STRIDE);
    v16h b3 = *(const v16h*)(pb + 3 * B_NT_STRIDE);

    for (int kt = 0; kt < HID_ / 32 - 1; ++kt) {
        pa += 128;
        pb += B_KT_STRIDE;
        // Issue next iteration's loads first (overlap with the 4 WMMAs below).
        v4f g0 = *(const v4f*)(pa +  0);
        v4f g1 = *(const v4f*)(pa + 16);
        v4f g2 = *(const v4f*)(pa + 64);
        v4f g3 = *(const v4f*)(pa + 80);
        v16h n0 = *(const v16h*)(pb + 0 * B_NT_STRIDE);
        v16h n1 = *(const v16h*)(pb + 1 * B_NT_STRIDE);
        v16h n2 = *(const v16h*)(pb + 2 * B_NT_STRIDE);
        v16h n3 = *(const v16h*)(pb + 3 * B_NT_STRIDE);

        v16h a;
        #pragma unroll
        for (int j = 0; j < 4; ++j) {
            a[j]      = (half_t)f0[j];
            a[4 + j]  = (half_t)f1[j];
            a[8 + j]  = (half_t)f2[j];
            a[12 + j] = (half_t)f3[j];
        }
        acc[0] = __builtin_amdgcn_wmma_f32_16x16x32_f16(false, a, false, b0, (short)0, acc[0], false, false);
        acc[1] = __builtin_amdgcn_wmma_f32_16x16x32_f16(false, a, false, b1, (short)0, acc[1], false, false);
        acc[2] = __builtin_amdgcn_wmma_f32_16x16x32_f16(false, a, false, b2, (short)0, acc[2], false, false);
        acc[3] = __builtin_amdgcn_wmma_f32_16x16x32_f16(false, a, false, b3, (short)0, acc[3], false, false);

        f0 = g0; f1 = g1; f2 = g2; f3 = g3;
        b0 = n0; b1 = n1; b2 = n2; b3 = n3;
    }
    {   // Pipeline epilogue: last K-tile.
        v16h a;
        #pragma unroll
        for (int j = 0; j < 4; ++j) {
            a[j]      = (half_t)f0[j];
            a[4 + j]  = (half_t)f1[j];
            a[8 + j]  = (half_t)f2[j];
            a[12 + j] = (half_t)f3[j];
        }
        acc[0] = __builtin_amdgcn_wmma_f32_16x16x32_f16(false, a, false, b0, (short)0, acc[0], false, false);
        acc[1] = __builtin_amdgcn_wmma_f32_16x16x32_f16(false, a, false, b1, (short)0, acc[1], false, false);
        acc[2] = __builtin_amdgcn_wmma_f32_16x16x32_f16(false, a, false, b2, (short)0, acc[2], false, false);
        acc[3] = __builtin_amdgcn_wmma_f32_16x16x32_f16(false, a, false, b3, (short)0, acc[3], false, false);
    }

    // Epilogue: + bias, store f16 row-major. D(m,n): vgpr r -> m = r + hi16*8, col = lane&15.
    const int roff = hi16 * 8;
    #pragma unroll
    for (int i = 0; i < 4; ++i) {
        const int col = (ntb + i) * 16 + (lane & 15);
        const float bb = bv[col];
        #pragma unroll
        for (int r = 0; r < 8; ++r) {
            vrm[(size_t)(mt * 16 + roff + r) * NOUT_ + col] = (half_t)(acc[i][r] + bb);
        }
    }
}

// ---------------------------------------------------------------------------
// GEMM2: out = diag(inv) * attn16 @ value16, per batch, causal K-tile skip.
// Wave computes a 16x64 strip; same 2-stage pipeline as GEMM1.
// ---------------------------------------------------------------------------
__global__ void __launch_bounds__(256) gemm_attn_kernel(const half_t* __restrict__ a16,
                                                        const half_t* __restrict__ vpk,
                                                        const float* __restrict__ inv,
                                                        float* __restrict__ out) {
    const int lane = threadIdx.x & 31;
    const int wave = threadIdx.x >> 5;
    const int qt   = blockIdx.x;               // 0..127
    const int ntg  = blockIdx.y * 8 + wave;    // 0..15
    const int b    = blockIdx.z;               // 0..3
    const int ntb  = ntg * 4;
    const int hi16 = (lane < 16) ? 0 : 1;
    const int q    = qt * 16 + (lane & 15);

    // A: f16 row; per kt the lane reads 16B at +0 and +32 from
    // (row base + hi16*16 + kt*64).
    const char* pa = (const char*)(a16 + ((size_t)b * S_ + q) * S_) + hi16 * 16;
    // B: packed fragments, per-batch slab of 4 MB.
    const char* pb = (const char*)vpk + (size_t)b * ((size_t)64 * 64 * 32 * 32)
                   + (size_t)(ntb * 32 + lane) * 32;

    v8f acc[4] = {};
    const int ktmax = (qt * 16 + 15) / 32 + 1;   // causal: skip all-zero K tiles (>=1)

    // Prologue: stage kt = 0.
    v8h alo = *(const v8h*)(pa + 0);
    v8h ahi = *(const v8h*)(pa + 32);
    v16h b0 = *(const v16h*)(pb + 0 * B_NT_STRIDE);
    v16h b1 = *(const v16h*)(pb + 1 * B_NT_STRIDE);
    v16h b2 = *(const v16h*)(pb + 2 * B_NT_STRIDE);
    v16h b3 = *(const v16h*)(pb + 3 * B_NT_STRIDE);

    for (int kt = 0; kt < ktmax - 1; ++kt) {
        pa += 64;
        pb += B_KT_STRIDE;
        v8h nlo = *(const v8h*)(pa + 0);
        v8h nhi = *(const v8h*)(pa + 32);
        v16h n0 = *(const v16h*)(pb + 0 * B_NT_STRIDE);
        v16h n1 = *(const v16h*)(pb + 1 * B_NT_STRIDE);
        v16h n2 = *(const v16h*)(pb + 2 * B_NT_STRIDE);
        v16h n3 = *(const v16h*)(pb + 3 * B_NT_STRIDE);

        v16h a = __builtin_shufflevector(alo, ahi, 0, 1, 2, 3, 4, 5, 6, 7,
                                         8, 9, 10, 11, 12, 13, 14, 15);
        acc[0] = __builtin_amdgcn_wmma_f32_16x16x32_f16(false, a, false, b0, (short)0, acc[0], false, false);
        acc[1] = __builtin_amdgcn_wmma_f32_16x16x32_f16(false, a, false, b1, (short)0, acc[1], false, false);
        acc[2] = __builtin_amdgcn_wmma_f32_16x16x32_f16(false, a, false, b2, (short)0, acc[2], false, false);
        acc[3] = __builtin_amdgcn_wmma_f32_16x16x32_f16(false, a, false, b3, (short)0, acc[3], false, false);

        alo = nlo; ahi = nhi;
        b0 = n0; b1 = n1; b2 = n2; b3 = n3;
    }
    {   // Pipeline epilogue: last K-tile.
        v16h a = __builtin_shufflevector(alo, ahi, 0, 1, 2, 3, 4, 5, 6, 7,
                                         8, 9, 10, 11, 12, 13, 14, 15);
        acc[0] = __builtin_amdgcn_wmma_f32_16x16x32_f16(false, a, false, b0, (short)0, acc[0], false, false);
        acc[1] = __builtin_amdgcn_wmma_f32_16x16x32_f16(false, a, false, b1, (short)0, acc[1], false, false);
        acc[2] = __builtin_amdgcn_wmma_f32_16x16x32_f16(false, a, false, b2, (short)0, acc[2], false, false);
        acc[3] = __builtin_amdgcn_wmma_f32_16x16x32_f16(false, a, false, b3, (short)0, acc[3], false, false);
    }

    // Epilogue: scale each output row by inv[q_row], store f32.
    const int roff = hi16 * 8;
    const float* invp = inv + b * S_ + qt * 16 + roff;
    v4f s0 = *(const v4f*)invp;
    v4f s1 = *(const v4f*)(invp + 4);
    #pragma unroll
    for (int i = 0; i < 4; ++i) {
        const int col = (ntb + i) * 16 + (lane & 15);
        #pragma unroll
        for (int r = 0; r < 8; ++r) {
            const float sc = (r < 4) ? s0[r] : s1[r - 4];
            out[((size_t)b * S_ + qt * 16 + roff + r) * NOUT_ + col] = acc[i][r] * sc;
        }
    }
}

// ---------------------------------------------------------------------------
extern "C" void kernel_launch(void* const* d_in, const int* in_sizes, int n_in,
                              void* d_out, int out_size, void* d_ws, size_t ws_size,
                              hipStream_t stream) {
    const float* hid   = (const float*)d_in[0];   // [B,S,HID]
    const float* assoc = (const float*)d_in[1];   // [B,S,S]
    // d_in[2] = broad_hc_attn: unused (broad_heads == 0 in the reference)
    const float* Wv    = (const float*)d_in[3];   // [HID, NH*HD]
    const float* bv    = (const float*)d_in[4];   // [NH*HD]
    float*       out   = (float*)d_out;           // [B,S,NH*HD]

    char*   ws   = (char*)d_ws;
    half_t* wv16 = (half_t*)(ws + OFF_WV16);
    half_t* vrm  = (half_t*)(ws + OFF_VRM);
    half_t* vpk  = (half_t*)(ws + OFF_VPK);
    half_t* a16  = (half_t*)(ws + OFF_A16);
    float*  inv  = (float*)(ws + OFF_INV);

    // 1) Wv -> f16 B-fragment-packed layout
    repack_wv_kernel<<<(HID_ * NOUT_) / 256, 256, 0, stream>>>(Wv, wv16);
    // 2) causal mask + f16 convert + inverse row sums
    mask_rowsum_kernel<<<B_ * S_, 256, 0, stream>>>(assoc, a16, inv);
    // 3) value projection GEMM (f16 WMMA, f32 accum, +bias)
    gemm_value_kernel<<<dim3((B_ * S_) / 16, 2), 256, 0, stream>>>(hid, wv16, bv, vrm);
    // 4) value -> B-fragment-packed per batch
    repack_val_kernel<<<(B_ * S_ * NOUT_) / 256, 256, 0, stream>>>(vrm, vpk);
    // 5) attention GEMM with causal K-skip + row-norm epilogue
    gemm_attn_kernel<<<dim3(S_ / 16, 2, B_), 256, 0, stream>>>(a16, vpk, inv, out);
}
